// VelocityNetwork_87024627351631
// MI455X (gfx1250) — matrix operand order repaired and checked by reading, not compiled, not run on hardware
//
#include <hip/hip_runtime.h>
#include <math.h>

// ---------------------------------------------------------------------------
// VelocityNetwork (DiT, seq_len==1) for MI455X / gfx1250.
// All GEMMs run as bf16 V_WMMA_F32_16X16X32_BF16 with fp32 accumulation.
// LDS tiles are staged with GLOBAL_LOAD_ASYNC_TO_LDS_B128 (ASYNCcnt path),
// double-buffered so copies for tile k+1 overlap the WMMAs on tile k.
// Attention degenerates (softmax over 1 key == 1), so only the V projection
// of QKV is computed.
// ---------------------------------------------------------------------------

typedef __attribute__((ext_vector_type(16))) __bf16 v16bf;
typedef __attribute__((ext_vector_type(8)))  __bf16 v8bf;
typedef __attribute__((ext_vector_type(8)))  float  v8f;

// Async global -> LDS copy of 16 bytes. VDST operand = u32 LDS byte address,
// VADDR = 64-bit global address (GV mode). Tracked by ASYNCcnt.
__device__ __forceinline__ void async_copy_b128(uint32_t lds_addr, const void* gptr)
{
    asm volatile("global_load_async_to_lds_b128 %0, %1, off"
                 :: "v"(lds_addr), "v"(gptr)
                 : "memory");
}
__device__ __forceinline__ void wait_asynccnt0()
{
    asm volatile("s_wait_asynccnt 0x0" ::: "memory");
}

// Low 32 bits of a flat LDS pointer == LDS byte offset (ISA: LDS aperture
// truncates addr[31:0]).
__device__ __forceinline__ uint32_t lds_addr_of(const void* p)
{
    return (uint32_t)(uintptr_t)p;
}

// ---------------- fp32 -> bf16 conversion (grid-stride, float4) -------------
__global__ __launch_bounds__(256)
void cvt_f32_bf16(const float* __restrict__ src, __bf16* __restrict__ dst, size_t n)
{
    size_t i = (size_t)blockIdx.x * blockDim.x + threadIdx.x;
    size_t stride = (size_t)gridDim.x * blockDim.x;
    for (size_t p = i * 4; p < n; p += stride * 4) {
        float4 f = *(const float4*)(src + p);
        dst[p + 0] = (__bf16)f.x;
        dst[p + 1] = (__bf16)f.y;
        dst[p + 2] = (__bf16)f.z;
        dst[p + 3] = (__bf16)f.w;
    }
}

// ---------------- silu(t_emb) -> bf16 ---------------------------------------
__global__ __launch_bounds__(256)
void silu_bf16(const float* __restrict__ x, __bf16* __restrict__ y, int n)
{
    int i = blockIdx.x * blockDim.x + threadIdx.x;
    if (i < n) {
        float t = x[i];
        y[i] = (__bf16)(t / (1.0f + expf(-t)));
    }
}

// ---------------- fused LayerNorm (+ optional adaLN modulation) -------------
__global__ __launch_bounds__(256)
void ln_mod_bf16(const float* __restrict__ X,
                 const float* __restrict__ g, const float* __restrict__ b,
                 const float* __restrict__ ada, int ada_ld, int s_off, int sh_off,
                 __bf16* __restrict__ Y, int D)
{
    const int row  = blockIdx.x;
    const int tid  = threadIdx.x;
    const int lane = tid & 31;
    const int wave = tid >> 5;
    const float* x = X + (size_t)row * D;

    float v[4];
    float lsum = 0.0f, lsq = 0.0f;
#pragma unroll
    for (int i = 0; i < 4; ++i) {
        float t = x[tid + i * 256];
        v[i] = t;
        lsum += t;
        lsq  += t * t;
    }
#pragma unroll
    for (int o = 16; o > 0; o >>= 1) {
        lsum += __shfl_xor(lsum, o, 32);
        lsq  += __shfl_xor(lsq,  o, 32);
    }
    __shared__ float ssum[8], ssq[8];
    if (lane == 0) { ssum[wave] = lsum; ssq[wave] = lsq; }
    __syncthreads();
    if (wave == 0) {
        float a = (lane < 8) ? ssum[lane] : 0.0f;
        float q = (lane < 8) ? ssq[lane]  : 0.0f;
#pragma unroll
        for (int o = 4; o > 0; o >>= 1) {
            a += __shfl_xor(a, o, 32);
            q += __shfl_xor(q, o, 32);
        }
        if (lane == 0) { ssum[0] = a; ssq[0] = q; }
    }
    __syncthreads();
    const float mean = ssum[0] / (float)D;
    const float var  = ssq[0] / (float)D - mean * mean;
    const float rstd = rsqrtf(var + 1e-5f);
    const float* srow = ada ? (ada + (size_t)row * ada_ld) : nullptr;
#pragma unroll
    for (int i = 0; i < 4; ++i) {
        int c = tid + i * 256;
        float y = (v[i] - mean) * rstd * g[c] + b[c];
        if (srow) y = y * (1.0f + srow[s_off + c]) + srow[sh_off + c];
        Y[(size_t)row * D + c] = (__bf16)y;
    }
}

// ---------------- bf16 WMMA GEMM: C = A @ W^T + bias (+ epilogue) -----------
// A: [M,K] bf16 row-major (lda), W: [N,K] bf16 row-major (ldw).
// Block tile 128x128, K-step 32, double-buffered LDS. 8 waves in a 4x2 grid;
// each wave computes a 32x64 subtile = 2x4 v_wmma_f32_16x16x32_bf16 accums.
enum { EPI_F32 = 0, EPI_ADD_F32 = 1, EPI_BF16 = 2, EPI_GELU_BF16 = 3 };

template <int EPI>
__global__ __launch_bounds__(256)
void gemm_bf16_wmma(const __bf16* __restrict__ A, int lda,
                    const __bf16* __restrict__ W, int ldw,
                    const float* __restrict__ bias,
                    const float* __restrict__ extra, int lde,
                    void* __restrict__ Cout, int ldc,
                    int K)
{
    // [buf][row][k], +8 bf16 pad (16B) per row; one buffer = 10240 B.
    __shared__ __align__(16) __bf16 As[2][128][40];
    __shared__ __align__(16) __bf16 Bs[2][128][40];
    const uint32_t BUF = 128 * 40 * 2;   // bytes per buffer

    const int tid  = threadIdx.x;
    const int lane = tid & 31;
    const int wave = tid >> 5;
    const int wm   = wave & 3;    // 4 waves along M (32 rows each)
    const int wn   = wave >> 2;   // 2 waves along N (64 cols each)
    const size_t bm = (size_t)blockIdx.y * 128;
    const size_t bn = (size_t)blockIdx.x * 128;

    v8f acc[2][4];
#pragma unroll
    for (int i = 0; i < 2; ++i)
#pragma unroll
        for (int j = 0; j < 4; ++j)
#pragma unroll
            for (int r = 0; r < 8; ++r) acc[i][j][r] = 0.0f;

    // Per-thread staging: 512 chunks of 8 bf16 per 128x32 tile; this thread
    // owns chunks {tid, tid+256} of each of As / Bs.
    const int r0  = tid >> 2;                 // rows 0..63   (chunk 0)
    const int r1  = r0 + 64;                  // rows 64..127 (chunk 1)
    const int c8  = (tid & 3) << 3;           // K sub-offset 0/8/16/24
    const uint32_t asA0 = lds_addr_of(&As[0][r0][c8]);
    const uint32_t asA1 = lds_addr_of(&As[0][r1][c8]);
    const uint32_t asB0 = lds_addr_of(&Bs[0][r0][c8]);
    const uint32_t asB1 = lds_addr_of(&Bs[0][r1][c8]);
    const __bf16* gA0 = A + (bm + r0) * (size_t)lda + c8;
    const __bf16* gA1 = A + (bm + r1) * (size_t)lda + c8;
    const __bf16* gB0 = W + (bn + r0) * (size_t)ldw + c8;
    const __bf16* gB1 = W + (bn + r1) * (size_t)ldw + c8;

    auto stage = [&](int buf, int k0) {
        const uint32_t o = (uint32_t)buf * BUF;
        async_copy_b128(asA0 + o, gA0 + k0);
        async_copy_b128(asA1 + o, gA1 + k0);
        async_copy_b128(asB0 + o, gB0 + k0);
        async_copy_b128(asB1 + o, gB1 + k0);
    };

    stage(0, 0);                       // prologue: fill buffer 0
    const int nk = K >> 5;             // K is a multiple of 32
    for (int it = 0; it < nk; ++it) {
        const int cur = it & 1;
        wait_asynccnt0();              // my copies into `cur` have landed
        __syncthreads();               // everyone's landed; cur^1 readers done
        const int kn = (it + 1) << 5;
        if (kn < K) {
            stage(cur ^ 1, kn);        // overlaps with the WMMAs below
            if (kn + 32 < K) {         // warm L2 two tiles ahead
                __builtin_prefetch(gA0 + kn + 32, 0, 2);
                __builtin_prefetch(gB0 + kn + 32, 0, 2);
            }
        }

        // Fragment assembly per ISA 7.12.2 layouts.
        const int rl = lane & 15;
        const int kh = (lane >> 4) << 3;   // A half-wave K base (interleaved 8s)
    const int kb = (lane >> 4) << 3;   // B half-wave K base (interleaved 8s)

        v16bf af[2], bfv[4];
#pragma unroll
        for (int i = 0; i < 2; ++i) {
            const __bf16* p = &As[cur][wm * 32 + i * 16 + rl][kh];
            v8bf lo = *(const v8bf*)p;          // K = kh .. kh+7
            v8bf hi = *(const v8bf*)(p + 16);   // K = kh+16 .. kh+23
            af[i] = __builtin_shufflevector(lo, hi,
                     0,1,2,3,4,5,6,7,8,9,10,11,12,13,14,15);
        }
#pragma unroll
        for (int j = 0; j < 4; ++j) {
            const __bf16* p = &Bs[cur][wn * 64 + j * 16 + rl][kb];
            v8bf lo = *(const v8bf*)p;          // K = kb .. kb+7
            v8bf hi = *(const v8bf*)(p + 16);   // K = kb+16 .. kb+23
            bfv[j] = __builtin_shufflevector(lo, hi,
                     0,1,2,3,4,5,6,7,8,9,10,11,12,13,14,15);
        }
#pragma unroll
        for (int i = 0; i < 2; ++i)
#pragma unroll
            for (int j = 0; j < 4; ++j)
                acc[i][j] = __builtin_amdgcn_wmma_f32_16x16x32_bf16(
                    false, af[i], false, bfv[j], (short)0, acc[i][j], false, false);
    }

    // Epilogue. C/D layout: VGPR r, lane l -> M = 8*(l>>4)+r, N = (l&15).
    const int cr = lane & 15;
    const int ch = lane >> 4;
#pragma unroll
    for (int i = 0; i < 2; ++i) {
#pragma unroll
        for (int j = 0; j < 4; ++j) {
            const size_t ncol = bn + wn * 64 + j * 16 + cr;
            const float bval = bias[ncol];
#pragma unroll
            for (int r = 0; r < 8; ++r) {
                const size_t mrow = bm + wm * 32 + i * 16 + ch * 8 + r;
                float c = acc[i][j][r] + bval;
                if (EPI == EPI_ADD_F32) {
                    ((float*)Cout)[mrow * ldc + ncol] = c + extra[mrow * lde + ncol];
                } else if (EPI == EPI_F32) {
                    ((float*)Cout)[mrow * ldc + ncol] = c;
                } else if (EPI == EPI_BF16) {
                    ((__bf16*)Cout)[mrow * ldc + ncol] = (__bf16)c;
                } else { // EPI_GELU_BF16 (exact erf GELU)
                    float gv = 0.5f * c * (1.0f + erff(c * 0.70710678118654752f));
                    ((__bf16*)Cout)[mrow * ldc + ncol] = (__bf16)gv;
                }
            }
        }
    }
}

// ---------------------------------------------------------------------------
extern "C" void kernel_launch(void* const* d_in, const int* in_sizes, int n_in,
                              void* d_out, int out_size, void* d_ws, size_t ws_size,
                              hipStream_t stream)
{
    (void)in_sizes; (void)n_in; (void)out_size; (void)ws_size;
    const int B = 4096, D = 1024, TE = 256, L = 4;

    const float* x      = (const float*)d_in[0];
    const float* t_emb  = (const float*)d_in[1];
    const float* cond   = (const float*)d_in[2];
    const float* in_w   = (const float*)d_in[3];
    const float* in_b   = (const float*)d_in[4];
    const float* ln1_g  = (const float*)d_in[5];
    const float* ln1_b  = (const float*)d_in[6];
    const float* wqkv   = (const float*)d_in[7];
    const float* bqkv   = (const float*)d_in[8];
    const float* wo     = (const float*)d_in[9];
    const float* bo     = (const float*)d_in[10];
    const float* ln2_g  = (const float*)d_in[11];
    const float* ln2_b  = (const float*)d_in[12];
    const float* w1     = (const float*)d_in[13];
    const float* b1     = (const float*)d_in[14];
    const float* w2     = (const float*)d_in[15];
    const float* b2     = (const float*)d_in[16];
    const float* ada_w  = (const float*)d_in[17];
    const float* ada_b  = (const float*)d_in[18];
    const float* oln_g  = (const float*)d_in[19];
    const float* oln_b  = (const float*)d_in[20];
    const float* out_w  = (const float*)d_in[21];
    const float* out_b  = (const float*)d_in[22];

    // ---- workspace carve-up (256B aligned) ----
    char* ws = (char*)d_ws;
    size_t off = 0;
    auto alloc = [&](size_t bytes) -> char* {
        char* p = ws + off;
        off += (bytes + 255) & ~(size_t)255;
        return p;
    };
    __bf16* x16   = (__bf16*)alloc((size_t)B * D * 2);
    __bf16* st16  = (__bf16*)alloc((size_t)B * TE * 2);
    float*  h     = (float*) alloc((size_t)B * D * 4);
    float*  ada   = (float*) alloc((size_t)B * 4 * D * 4);
    __bf16* xn16  = (__bf16*)alloc((size_t)B * D * 2);
    __bf16* v16   = (__bf16*)alloc((size_t)B * D * 2);
    __bf16* m116  = (__bf16*)alloc((size_t)B * 4 * D * 2);
    __bf16* inw16 = (__bf16*)alloc((size_t)D * D * 2);
    __bf16* ow16  = (__bf16*)alloc((size_t)D * D * 2);
    __bf16 *wv16[4], *wo16[4], *w116[4], *w216[4], *aw16[4];
    for (int l = 0; l < L; ++l) {
        wv16[l] = (__bf16*)alloc((size_t)D * D * 2);
        wo16[l] = (__bf16*)alloc((size_t)D * D * 2);
        w116[l] = (__bf16*)alloc((size_t)4 * D * D * 2);
        w216[l] = (__bf16*)alloc((size_t)4 * D * D * 2);
        aw16[l] = (__bf16*)alloc((size_t)4 * D * TE * 2);
    }

    auto cvt = [&](const float* s, __bf16* d, size_t n) {
        cvt_f32_bf16<<<1024, 256, 0, stream>>>(s, d, n);
    };

    // ---- one-time-per-call precision downcast of operands ----
    cvt(x, x16, (size_t)B * D);
    silu_bf16<<<(B * TE) / 256, 256, 0, stream>>>(t_emb, st16, B * TE);
    cvt(in_w, inw16, (size_t)D * D);
    cvt(out_w, ow16, (size_t)D * D);
    for (int l = 0; l < L; ++l) {
        // only the V slice of wqkv is ever needed (softmax over 1 key == 1)
        cvt(wqkv + ((size_t)l * 3 * D + 2 * D) * D, wv16[l], (size_t)D * D);
        cvt(wo + (size_t)l * D * D,        wo16[l], (size_t)D * D);
        cvt(w1 + (size_t)l * 4 * D * D,    w116[l], (size_t)4 * D * D);
        cvt(w2 + (size_t)l * 4 * D * D,    w216[l], (size_t)4 * D * D);
        cvt(ada_w + (size_t)l * 4 * D * TE, aw16[l], (size_t)4 * D * TE);
    }

    const dim3 blk(256);
    const dim3 gD (D / 128,     B / 128);   // N=1024
    const dim3 g4D(4 * D / 128, B / 128);   // N=4096

    // h = x @ in_w^T + in_b + condition
    gemm_bf16_wmma<EPI_ADD_F32><<<gD, blk, 0, stream>>>(
        x16, D, inw16, D, in_b, cond, D, h, D, D);

    for (int l = 0; l < L; ++l) {
        // ada = silu(t_emb) @ ada_w^T + ada_b           [B, 4D] fp32
        gemm_bf16_wmma<EPI_F32><<<g4D, blk, 0, stream>>>(
            st16, TE, aw16[l], TE, ada_b + (size_t)l * 4 * D, nullptr, 0, ada, 4 * D, TE);
        // xn = LN1(h)*(1+s1)+sh1                        -> bf16
        ln_mod_bf16<<<B, blk, 0, stream>>>(
            h, ln1_g + (size_t)l * D, ln1_b + (size_t)l * D, ada, 4 * D, 0, D, xn16, D);
        // v  = xn @ Wv^T + bv                           -> bf16
        gemm_bf16_wmma<EPI_BF16><<<gD, blk, 0, stream>>>(
            xn16, D, wv16[l], D, bqkv + (size_t)l * 3 * D + 2 * D, nullptr, 0, v16, D, D);
        // h += v @ wo^T + bo                            (attn==identity)
        gemm_bf16_wmma<EPI_ADD_F32><<<gD, blk, 0, stream>>>(
            v16, D, wo16[l], D, bo + (size_t)l * D, h, D, h, D, D);
        // xn = LN2(h)*(1+s2)+sh2                        -> bf16
        ln_mod_bf16<<<B, blk, 0, stream>>>(
            h, ln2_g + (size_t)l * D, ln2_b + (size_t)l * D, ada, 4 * D, 2 * D, 3 * D, xn16, D);
        // m1 = gelu(xn @ w1^T + b1)                     -> bf16
        gemm_bf16_wmma<EPI_GELU_BF16><<<g4D, blk, 0, stream>>>(
            xn16, D, w116[l], D, b1 + (size_t)l * 4 * D, nullptr, 0, m116, 4 * D, D);
        // h += m1 @ w2^T + b2
        gemm_bf16_wmma<EPI_ADD_F32><<<gD, blk, 0, stream>>>(
            m116, 4 * D, w216[l], 4 * D, b2 + (size_t)l * D, h, D, h, D, 4 * D);
    }

    // out = LN(h) @ out_w^T + out_b
    ln_mod_bf16<<<B, blk, 0, stream>>>(h, oln_g, oln_b, nullptr, 0, 0, 0, xn16, D);
    gemm_bf16_wmma<EPI_F32><<<gD, blk, 0, stream>>>(
        xn16, D, ow16, D, out_b, nullptr, 0, (float*)d_out, D, D);
}
